// VQVAE_39556648796222
// MI455X (gfx1250) — matrix operand order repaired and checked by reading, not compiled, not run on hardware
//
#include <hip/hip_runtime.h>
#include <hip/hip_bf16.h>

// ---------------------------------------------------------------------------
// CDNA5 (gfx1250) VQ-VAE forward.
// Implicit-GEMM 3D conv on v_wmma_f32_16x16x32_f16, wave32.
// Block = 128 threads (4 waves). Wave tile: 16(M spatial) x 64(N=Cout) via 4
// accumulators reusing one A fragment. LDS fragments double-buffered; staging
// is branchless, uses packed v_cvt_pk_rtz_f16_f32 + 32-bit cndmask; global
// prefetch of chunk k+1 overlaps the WMMAs of chunk k; one barrier per step.
// ---------------------------------------------------------------------------

typedef _Float16 v16h __attribute__((ext_vector_type(16)));
typedef _Float16 h8   __attribute__((ext_vector_type(8)));
typedef _Float16 h2   __attribute__((ext_vector_type(2)));
typedef float    v8f  __attribute__((ext_vector_type(8)));

// Pack two fp32 -> one 32-bit packed f16 pair (v_cvt_pk_rtz_f16_f32).
__device__ __forceinline__ unsigned pk2(float a, float b) {
    return __builtin_bit_cast(unsigned, __builtin_amdgcn_cvt_pkrtz(a, b));
}

// Fragment layouts used here (wave32; contiguous k, not the ISA table order):
//  A 16x32 f16 : lane (g=lane>>4, r=lane&15) holds row M=r,
//                element i -> K = i + 16*g
//  B 32x16 f16 : lane holds col N=lane&15, element i -> K = i + 16*(lane>>4)
//  C/D 16x16 f32: VGPR rr -> M = rr + 8*(lane>>4), N = lane&15

template <int K3, int CIN, int STRIDE, int UP, int PAD, int LSI, int LSO, int NT>
__global__ __launch_bounds__(128)
void conv3d_wmma(const float* __restrict__ in, const float* __restrict__ wgt,
                 const float* __restrict__ bias, float* __restrict__ out,
                 int Cout,
                 const float* __restrict__ bn_g, const float* __restrict__ bn_b,
                 const float* __restrict__ pr, const float* __restrict__ res,
                 int relu_out)
{
    constexpr int KT  = K3 * K3 * K3 * CIN;      // GEMM-K total
    static_assert(KT % 32 == 0, "K must be a multiple of 32");
    constexpr int NK  = KT / 32;                 // K chunks
    constexpr int SI  = 1 << LSI;                // cubic input size
    constexpr int SO  = 1 << LSO;                // cubic output size
    constexpr int UPS = (UP == 2) ? 1 : 0;
    constexpr int TN  = 16 * NT;                 // N columns per block
    constexpr int RPT = 4 * NT;                  // B k-rows per thread

    __shared__ __align__(32) _Float16 Af[2][4][32][16];  // double-buffered A
    __shared__ __align__(32) _Float16 Bf[2][32][TN];     // double-buffered B

    const int tid   = threadIdx.x;
    const int w     = tid >> 5;                  // wave id 0..3
    const int lane  = tid & 31;
    const int mBase = (blockIdx.x * 4 + w) * 16;
    const int nBase = blockIdx.y * TN;

    // ---- A staging role: lane stages row arow, cols acolh*16 .. +15
    const int arow  = lane >> 1;
    const int acolh = lane & 1;
    const int m     = mBase + arow;
    const int ow    = m & (SO - 1);
    const int oh    = (m >> LSO) & (SO - 1);
    const int od    = m >> (2 * LSO);
    const int wn0   = ow * STRIDE - PAD;
    const int hn0   = oh * STRIDE - PAD;
    const int dn0   = od * STRIDE - PAD;

    // ---- B staging role: thread loads RPT contiguous k-rows for one n col
    const int bncol = tid % TN;
    const int bkr0  = (tid / TN) * RPT;
    const int bl    = (bncol & 15) + ((bkr0 >> 4) << 4);
    const int bio   = ((bncol >> 4) << 4) + (bkr0 & 15);
    const int bn    = nBase + bncol;
    const bool bok  = (bn < Cout);
    const int bnc   = bok ? bn : 0;

    auto clampi = [](int x, int lim) {
        return x < 0 ? 0 : (x >= lim ? lim - 1 : x);
    };

    // Branchless A chunk load -> packed f16 fragment halves.
    auto loadA = [&](int kBase, h8& lo, h8& hi) {
        union { h8 v; unsigned u[4]; } L, H;
        if (CIN >= 32) {
            // chunk lies within one kernel tap (32 | CIN): scalar tap decode
            const int tap  = kBase / CIN;
            const int cin0 = (kBase % CIN) + acolh * 16;
            const int kw = tap % K3, kh = (tap / K3) % K3, kd = tap / (K3 * K3);
            const int wn = wn0 + kw, hn = hn0 + kh, dn = dn0 + kd;
            const bool ok = ((((dn | hn | wn) & (UP - 1)) == 0) &&
                             dn >= 0 && (dn >> UPS) < SI &&
                             hn >= 0 && (hn >> UPS) < SI &&
                             wn >= 0 && (wn >> UPS) < SI);
            const int id = clampi(dn >> UPS, SI);
            const int ih = clampi(hn >> UPS, SI);
            const int iw = clampi(wn >> UPS, SI);
            const long sidx = ((((long)id << LSI) + ih) << LSI) + iw;
            const float* p  = in + sidx * CIN + cin0;
            float f[16];
#pragma unroll
            for (int j = 0; j < 16; ++j) f[j] = p[j];
#pragma unroll
            for (int j = 0; j < 4; ++j) {
                const unsigned ulo = pk2(f[2 * j], f[2 * j + 1]);
                const unsigned uhi = pk2(f[8 + 2 * j], f[9 + 2 * j]);
                L.u[j] = ok ? ulo : 0u;
                H.u[j] = ok ? uhi : 0u;
            }
        } else {
            // generic per-column path (enc0: CIN==1, K3=4 -> shifts only)
            float f[16];
#pragma unroll
            for (int j = 0; j < 16; ++j) {
                const int c   = acolh * 16 + j;
                const int kk  = kBase + c;
                const int tap = kk / CIN;
                const int cin = kk % CIN;
                const int kw = tap % K3, kh = (tap / K3) % K3, kd = tap / (K3 * K3);
                const int wn = wn0 + kw, hn = hn0 + kh, dn = dn0 + kd;
                const bool ok = ((((dn | hn | wn) & (UP - 1)) == 0) &&
                                 dn >= 0 && (dn >> UPS) < SI &&
                                 hn >= 0 && (hn >> UPS) < SI &&
                                 wn >= 0 && (wn >> UPS) < SI);
                const int id = clampi(dn >> UPS, SI);
                const int ih = clampi(hn >> UPS, SI);
                const int iw = clampi(wn >> UPS, SI);
                const long sidx = ((((long)id << LSI) + ih) << LSI) + iw;
                const float t = in[sidx * CIN + cin];
                f[j] = ok ? t : 0.0f;
            }
#pragma unroll
            for (int j = 0; j < 4; ++j) {
                L.u[j] = pk2(f[2 * j], f[2 * j + 1]);
                H.u[j] = pk2(f[8 + 2 * j], f[9 + 2 * j]);
            }
        }
        lo = L.v;
        hi = H.v;
    };

    // B chunk load -> packed f16 pairs (as 32-bit units).
    auto loadB = [&](int kBase, unsigned* tb) {
        float v[RPT];
#pragma unroll
        for (int j = 0; j < RPT; ++j)
            v[j] = wgt[(long)(kBase + bkr0 + j) * Cout + bnc];
#pragma unroll
        for (int j = 0; j < RPT / 2; ++j) {
            const unsigned u = pk2(v[2 * j], v[2 * j + 1]);
            tb[j] = bok ? u : 0u;
        }
    };

    auto storeA = [&](int p, h8 lo, h8 hi) {
        // cols c (=acolh*16+j): elem j of lane arow+16*acolh
        *(h8*)&Af[p][w][arow + 16 * acolh][0] = lo;
        *(h8*)&Af[p][w][arow + 16 * acolh][8] = hi;
    };
    auto storeB = [&](int p, const unsigned* tb) {
#pragma unroll
        for (int j = 0; j < RPT / 2; ++j)
            *(h2*)&Bf[p][bl][bio + 2 * j] = __builtin_bit_cast(h2, tb[j]);
    };

    v8f acc[NT];
#pragma unroll
    for (int nt = 0; nt < NT; ++nt) acc[nt] = (v8f){};

    auto compute = [&](int p) {
        const v16h a = *(const v16h*)&Af[p][w][lane][0];
#pragma unroll
        for (int nt = 0; nt < NT; ++nt) {
            const v16h b = *(const v16h*)&Bf[p][lane][nt * 16];
            acc[nt] = __builtin_amdgcn_wmma_f32_16x16x32_f16(
                false, a, false, b, (short)0, acc[nt], false, false);
        }
    };

    // -------- software pipeline: one barrier per K chunk --------
    {
        h8 lo = {}, hi = {};
        unsigned tb[RPT / 2];
        loadA(0, lo, hi);
        loadB(0, tb);
        storeA(0, lo, hi);
        storeB(0, tb);
        __syncthreads();
        for (int ks = 0; ks < NK - 1; ++ks) {
            const int p = ks & 1;
            loadA((ks + 1) * 32, lo, hi);   // global prefetch of next chunk
            loadB((ks + 1) * 32, tb);
            compute(p);                      // WMMAs hide prefetch latency
            storeA(p ^ 1, lo, hi);
            storeB(p ^ 1, tb);
            __syncthreads();
        }
        compute((NK - 1) & 1);
    }

    // -------- fused epilogue --------
    const int gl = lane >> 4;
    const int nl = lane & 15;
#pragma unroll
    for (int nt = 0; nt < NT; ++nt) {
        const int n = nBase + nt * 16 + nl;
        if (n < Cout) {
            const float bias_v = bias[n];
            const float gv = bn_g ? bn_g[n] : 1.0f;
            const float bv = bn_b ? bn_b[n] : 0.0f;
            const float av = pr ? pr[n] : 0.0f;
#pragma unroll
            for (int rr = 0; rr < 8; ++rr) {
                const long mm = (long)mBase + rr + gl * 8;
                float v = acc[nt][rr] + bias_v;
                if (bn_g) v = v * gv + bv;
                if (pr)   v = fmaxf(v, 0.0f) + av * fminf(v, 0.0f);
                if (res)  v += res[mm * Cout + n];
                if (relu_out) v = fmaxf(v, 0.0f);
                out[mm * Cout + n] = v;
            }
        }
    }
}

// ---------------------------------------------------------------------------
// Codebook norms: cnorm[j] = sum_d cb[j][d]^2   (512 codes x 64 dims)
// ---------------------------------------------------------------------------
__global__ __launch_bounds__(64)
void cnorm_kernel(const float* __restrict__ cb, float* __restrict__ cnorm)
{
    const int j = blockIdx.x * 64 + threadIdx.x;
    if (j < 512) {
        float s = 0.0f;
#pragma unroll
        for (int d = 0; d < 64; ++d) {
            const float c = cb[j * 64 + d];
            s = fmaf(c, c, s);
        }
        cnorm[j] = s;
    }
}

// ---------------------------------------------------------------------------
// VQ: per z-row argmin_j (||c_j||^2 - 2 z.c_j), write gathered codebook row.
// ---------------------------------------------------------------------------
__global__ __launch_bounds__(256)
void vq_kernel(const float* __restrict__ z, const float* __restrict__ cb,
               const float* __restrict__ cnorm, float* __restrict__ zq, int N)
{
    __shared__ float cbs[128][64];   // 32 KB codebook chunk
    const int row = blockIdx.x * 256 + threadIdx.x;

    float zr[64];
    if (row < N) {
#pragma unroll
        for (int d = 0; d < 64; ++d) zr[d] = z[(long)row * 64 + d];
    }

    float best = 3.0e38f;
    int   bi   = 0;
    for (int c0 = 0; c0 < 512; c0 += 128) {
        for (int e = threadIdx.x; e < 128 * 64; e += 256)
            cbs[e >> 6][e & 63] = cb[(long)(c0 + (e >> 6)) * 64 + (e & 63)];
        __syncthreads();
        if (row < N) {
            for (int j = 0; j < 128; ++j) {
                float dot = 0.0f;
#pragma unroll
                for (int d = 0; d < 64; ++d) dot = fmaf(zr[d], cbs[j][d], dot);
                const float dist = cnorm[c0 + j] - 2.0f * dot;
                if (dist < best) { best = dist; bi = c0 + j; }
            }
        }
        __syncthreads();
    }
    if (row < N) {
#pragma unroll
        for (int d = 0; d < 64; ++d) zq[(long)row * 64 + d] = cb[(long)bi * 64 + d];
    }
}

// ---------------------------------------------------------------------------
// Host launcher
// ---------------------------------------------------------------------------
extern "C" void kernel_launch(void* const* d_in, const int* in_sizes, int n_in,
                              void* d_out, int out_size, void* d_ws, size_t ws_size,
                              hipStream_t stream)
{
    (void)in_sizes; (void)n_in; (void)out_size; (void)ws_size;

    const float* x          = (const float*)d_in[0];
    const float* enc0_w     = (const float*)d_in[1];
    const float* enc0_b     = (const float*)d_in[2];
    const float* e0_c1_w    = (const float*)d_in[3];
    const float* e0_c1_b    = (const float*)d_in[4];
    const float* e0_c2_w    = (const float*)d_in[5];
    const float* e0_c2_b    = (const float*)d_in[6];
    const float* e0_bn_g    = (const float*)d_in[7];
    const float* e0_bn_b    = (const float*)d_in[8];
    const float* e0_pr      = (const float*)d_in[9];
    const float* enc1_w     = (const float*)d_in[10];
    const float* enc1_b     = (const float*)d_in[11];
    const float* e1_c1_w    = (const float*)d_in[12];
    const float* e1_c1_b    = (const float*)d_in[13];
    const float* e1_c2_w    = (const float*)d_in[14];
    const float* e1_c2_b    = (const float*)d_in[15];
    const float* e1_bn_g    = (const float*)d_in[16];
    const float* e1_bn_b    = (const float*)d_in[17];
    const float* e1_pr      = (const float*)d_in[18];
    const float* enc_out_w  = (const float*)d_in[19];
    const float* enc_out_b  = (const float*)d_in[20];
    const float* enc_pr     = (const float*)d_in[21];
    const float* codebook   = (const float*)d_in[22];
    const float* dec0_w     = (const float*)d_in[23];
    const float* dec0_b     = (const float*)d_in[24];
    const float* dec_pr     = (const float*)d_in[25];
    const float* d0_c1_w    = (const float*)d_in[26];
    const float* d0_c1_b    = (const float*)d_in[27];
    const float* d0_c2_w    = (const float*)d_in[28];
    const float* d0_c2_b    = (const float*)d_in[29];
    const float* d0_bn_g    = (const float*)d_in[30];
    const float* d0_bn_b    = (const float*)d_in[31];
    const float* d0_pr      = (const float*)d_in[32];
    const float* up0_w      = (const float*)d_in[33];
    const float* up0_b      = (const float*)d_in[34];
    const float* d1_c1_w    = (const float*)d_in[35];
    const float* d1_c1_b    = (const float*)d_in[36];
    const float* d1_c2_w    = (const float*)d_in[37];
    const float* d1_c2_b    = (const float*)d_in[38];
    const float* d1_bn_g    = (const float*)d_in[39];
    const float* d1_bn_b    = (const float*)d_in[40];
    const float* d1_pr      = (const float*)d_in[41];
    const float* up1_w      = (const float*)d_in[42];
    const float* up1_b      = (const float*)d_in[43];

    // Workspace: 3 ping-pong activation buffers (64 MiB each) + codebook norms
    const size_t BUF = 64ull * 64 * 64 * 64 * sizeof(float);
    float* A  = (float*)((char*)d_ws);
    float* B  = (float*)((char*)d_ws + BUF);
    float* C  = (float*)((char*)d_ws + 2 * BUF);
    float* CN = (float*)((char*)d_ws + 3 * BUF);

    float* out = (float*)d_out;
    const dim3 blk(128);
    const long M64  = 64L * 64 * 64;
    const long M32  = 32L * 32 * 32;
    const long M128 = 128L * 128 * 128;

    // ---------------- encoder ----------------
    conv3d_wmma<4, 1, 2, 1, 1, 7, 6, 4><<<dim3(M64 / 64, 1), blk, 0, stream>>>(
        x, enc0_w, enc0_b, A, 64, nullptr, nullptr, nullptr, nullptr, 1);
    conv3d_wmma<3, 64, 1, 1, 1, 6, 6, 4><<<dim3(M64 / 64, 1), blk, 0, stream>>>(
        A, e0_c1_w, e0_c1_b, B, 64, nullptr, nullptr, nullptr, nullptr, 1);
    conv3d_wmma<3, 64, 1, 1, 1, 6, 6, 4><<<dim3(M64 / 64, 1), blk, 0, stream>>>(
        B, e0_c2_w, e0_c2_b, C, 64, e0_bn_g, e0_bn_b, e0_pr, A, 1);
    conv3d_wmma<4, 64, 2, 1, 1, 6, 5, 4><<<dim3(M32 / 64, 2), blk, 0, stream>>>(
        C, enc1_w, enc1_b, A, 128, nullptr, nullptr, nullptr, nullptr, 1);
    conv3d_wmma<3, 128, 1, 1, 1, 5, 5, 4><<<dim3(M32 / 64, 2), blk, 0, stream>>>(
        A, e1_c1_w, e1_c1_b, B, 128, nullptr, nullptr, nullptr, nullptr, 1);
    conv3d_wmma<3, 128, 1, 1, 1, 5, 5, 4><<<dim3(M32 / 64, 2), blk, 0, stream>>>(
        B, e1_c2_w, e1_c2_b, C, 128, e1_bn_g, e1_bn_b, e1_pr, A, 1);
    conv3d_wmma<3, 128, 1, 1, 1, 5, 5, 4><<<dim3(M32 / 64, 1), blk, 0, stream>>>(
        C, enc_out_w, enc_out_b, A, 64, nullptr, nullptr, enc_pr, nullptr, 0);

    // ---------------- quantize ----------------
    cnorm_kernel<<<dim3(8), dim3(64), 0, stream>>>(codebook, CN);
    vq_kernel<<<dim3(32768 / 256), dim3(256), 0, stream>>>(A, codebook, CN, B, 32768);

    // ---------------- decoder ----------------
    conv3d_wmma<3, 64, 1, 1, 1, 5, 5, 4><<<dim3(M32 / 64, 2), blk, 0, stream>>>(
        B, dec0_w, dec0_b, C, 128, nullptr, nullptr, dec_pr, nullptr, 0);
    conv3d_wmma<3, 128, 1, 1, 1, 5, 5, 4><<<dim3(M32 / 64, 2), blk, 0, stream>>>(
        C, d0_c1_w, d0_c1_b, A, 128, nullptr, nullptr, nullptr, nullptr, 1);
    conv3d_wmma<3, 128, 1, 1, 1, 5, 5, 4><<<dim3(M32 / 64, 2), blk, 0, stream>>>(
        A, d0_c2_w, d0_c2_b, B, 128, d0_bn_g, d0_bn_b, d0_pr, C, 1);
    conv3d_wmma<4, 128, 1, 2, 2, 5, 6, 4><<<dim3(M64 / 64, 1), blk, 0, stream>>>(
        B, up0_w, up0_b, C, 64, nullptr, nullptr, nullptr, nullptr, 1);
    conv3d_wmma<3, 64, 1, 1, 1, 6, 6, 4><<<dim3(M64 / 64, 1), blk, 0, stream>>>(
        C, d1_c1_w, d1_c1_b, A, 64, nullptr, nullptr, nullptr, nullptr, 1);
    conv3d_wmma<3, 64, 1, 1, 1, 6, 6, 4><<<dim3(M64 / 64, 1), blk, 0, stream>>>(
        A, d1_c2_w, d1_c2_b, B, 64, d1_bn_g, d1_bn_b, d1_pr, C, 1);
    conv3d_wmma<4, 64, 1, 2, 2, 6, 7, 1><<<dim3(M128 / 64, 1), blk, 0, stream>>>(
        B, up1_w, up1_b, out, 1, nullptr, nullptr, nullptr, nullptr, 1);
}
